// PSABlock_EDFFN_11802570129738
// MI455X (gfx1250) — matrix operand order repaired and checked by reading, not compiled, not run on hardware
//
#include <hip/hip_runtime.h>
#include <math.h>

// ---------------------------------------------------------------------------
// PSABlock + EDFFN for MI455X (gfx1250, wave32, WMMA).
// GEMMs: v_wmma_f32_16x16x32_bf16 (fp32 accumulate).
// Spatial kernels: global_load_async_to_lds_b128 staging (ASYNCcnt path).
// ---------------------------------------------------------------------------

typedef __attribute__((ext_vector_type(16))) __bf16 v16bf;
typedef __attribute__((ext_vector_type(8)))  __bf16 v8bf;
typedef __attribute__((ext_vector_type(8)))  float  v8f;

#define WMMA_BF16(A, B, Cc) \
  __builtin_amdgcn_wmma_f32_16x16x32_bf16(false, (A), false, (B), (short)0, (Cc), false, false)

static constexpr int Bb   = 16;
static constexpr int C    = 256;
static constexpr int N    = 1024;   // H*W, H=W=32
static constexpr int NH   = 4;
static constexpr int HD   = 64;
static constexpr int KD   = 32;
static constexpr int HQKV = 512;    // C + 2*kd*NH
static constexpr int HID  = 512;    // 2*C

// ---- async global->LDS helpers (inline asm; portable across toolchains) ----
// LDS byte offset = low 32 bits of the generic pointer (flat LDS aperture:
// ISA 10.2).
__device__ __forceinline__ void async_load_b128(void* lds_dst, const void* gsrc) {
  unsigned l = (unsigned)(size_t)lds_dst;
  unsigned long long g = (unsigned long long)(size_t)gsrc;
  asm volatile("global_load_async_to_lds_b128 %0, %1, off" :: "v"(l), "v"(g) : "memory");
}
__device__ __forceinline__ void async_wait0() {
  asm volatile("s_wait_asynccnt 0" ::: "memory");
}

// ---------------------------------------------------------------------------
// Generic bf16 WMMA GEMM:  Out[b,m,n] = epi( sum_k Wt[m,k] * X[b,k,n] )
// Block tile: 64(M) x 128(N), K-step 32, 256 threads = 8 waves.
// LDS holds A/B tiles pre-arranged in WMMA fragment order:
//   A (16-bit 16x32): lane = (k>>4)*16 + (m&15), elem i = k&15
//   B (16-bit 32x16): lane = (k>>4)*16 + (n&15),      elem i = k&15
// ---------------------------------------------------------------------------
template <bool AFFINE, bool RES>
__global__ __launch_bounds__(256)
void gemm_bf16_kernel(const float* __restrict__ Wt, const float* __restrict__ X,
                      float* __restrict__ Out,
                      const float* __restrict__ Sc, const float* __restrict__ Bi,
                      const float* __restrict__ Res, int M, int K)
{
  __shared__ __align__(16) __bf16 sA[4 * 32 * 16];   // 4 m-subtiles
  __shared__ __align__(16) __bf16 sB[8 * 32 * 16];   // 8 n-subtiles

  const int tid  = threadIdx.x;
  const int lane = tid & 31;
  const int wv   = tid >> 5;
  const int b    = blockIdx.z;
  const int mbase = blockIdx.y * 64;
  const int nbase = blockIdx.x * 128;
  const int msub = wv & 3;
  const int ng0  = (wv >> 2) * 4;     // wave covers n-subtiles ng0..ng0+3

  v8f acc[4];
  #pragma unroll
  for (int t = 0; t < 4; ++t)
    #pragma unroll
    for (int j = 0; j < 8; ++j) acc[t][j] = 0.f;

  const int am  = tid >> 2;          // 0..63   (A loader row)
  const int akq = (tid & 3) * 8;     // 0,8,16,24 (A loader k-chunk)
  const int bn  = tid & 127;         // 0..127  (B loader column)
  const int bkh = tid >> 7;          // 0..1    (B loader k-half)

  for (int k0 = 0; k0 < K; k0 += 32) {
    // ---- stage A tile (64x32 fp32 -> bf16, fragment order) ----
    {
      const float* wp = Wt + (size_t)(mbase + am) * K + k0 + akq;
      float4 f0 = reinterpret_cast<const float4*>(wp)[0];
      float4 f1 = reinterpret_cast<const float4*>(wp)[1];
      v8bf p;
      p[0] = (__bf16)f0.x; p[1] = (__bf16)f0.y; p[2] = (__bf16)f0.z; p[3] = (__bf16)f0.w;
      p[4] = (__bf16)f1.x; p[5] = (__bf16)f1.y; p[6] = (__bf16)f1.z; p[7] = (__bf16)f1.w;
      const int lslot = (akq >> 4) * 16 + (am & 15);
      const int ib    = akq & 8;
      *reinterpret_cast<v8bf*>(&sA[((am >> 4) * 32 + lslot) * 16 + ib]) = p;
    }
    // ---- stage B tile (32x128 fp32 -> bf16, fragment order) ----
    {
      const float* xp = X + ((size_t)b * K + k0 + bkh * 16) * N + nbase + bn;
      v8bf p0, p1;
      #pragma unroll
      for (int j = 0; j < 8; ++j)  p0[j] = (__bf16)xp[(size_t)j * N];
      #pragma unroll
      for (int j = 0; j < 8; ++j)  p1[j] = (__bf16)xp[(size_t)(j + 8) * N];
      const int base = ((bn >> 4) * 32 + bkh * 16 + (bn & 15)) * 16;
      *reinterpret_cast<v8bf*>(&sB[base])     = p0;
      *reinterpret_cast<v8bf*>(&sB[base + 8]) = p1;
    }
    __syncthreads();

    const v16bf af = *reinterpret_cast<const v16bf*>(&sA[(msub * 32 + lane) * 16]);
    #pragma unroll
    for (int t = 0; t < 4; ++t) {
      const v16bf bfr = *reinterpret_cast<const v16bf*>(&sB[((ng0 + t) * 32 + lane) * 16]);
      acc[t] = WMMA_BF16(af, bfr, acc[t]);
    }
    __syncthreads();
  }

  // ---- epilogue: C-layout row = j + (lane>>4)*8, col = lane&15 ----
  const int hi = lane >> 4, col = lane & 15;
  #pragma unroll
  for (int t = 0; t < 4; ++t) {
    #pragma unroll
    for (int j = 0; j < 8; ++j) {
      const int m = mbase + msub * 16 + j + hi * 8;
      const int n = nbase + (ng0 + t) * 16 + col;
      float v = acc[t][j];
      if (AFFINE) v = v * Sc[m] + Bi[m];
      const size_t o = ((size_t)b * M + m) * N + n;
      if (RES) v += Res[o];
      Out[o] = v;
    }
  }
}

// ---------------------------------------------------------------------------
// Flash attention: per (b,h); each of 8 waves owns 16 query rows.
// scores = (q*scale)^T k  (K=kd=32 -> one bf16 WMMA per 16 keys),
// online softmax with wave32 shfl_xor row-reductions, probabilities
// bounced through LDS into A-fragment order, out += P * V^T (4 WMMAs / 32 keys).
// ---------------------------------------------------------------------------
__global__ __launch_bounds__(256)
void attn_kernel(const float* __restrict__ qkv, float* __restrict__ aout)
{
  __shared__ __align__(16) __bf16 pbuf[8 * 32 * 16];   // per-wave P tiles

  const int tid = threadIdx.x, lane = tid & 31, wv = tid >> 5;
  const int hi = lane >> 4, col = lane & 15;
  const int b = blockIdx.z, h = blockIdx.y;
  const int n0 = blockIdx.x * 128 + wv * 16;

  const float* qh = qkv + ((size_t)b * HQKV + h * 128) * N;
  const float* kh = qh + (size_t)KD * N;
  const float* vh = qh + (size_t)(2 * KD) * N;

  // Q as A-fragment (rows = queries, k = channel), scale folded in
  v16bf qf;
  #pragma unroll
  for (int i = 0; i < 16; ++i) {
    const int kk = ((i >> 3) << 4) | (hi << 3) | (i & 7);
    qf[i] = (__bf16)(qh[(size_t)kk * N + n0 + col] * 0.17677669529663687f);
  }

  float mrun[8], lrun[8];
  #pragma unroll
  for (int j = 0; j < 8; ++j) { mrun[j] = -1e30f; lrun[j] = 0.f; }
  v8f acco[4];
  #pragma unroll
  for (int t = 0; t < 4; ++t)
    #pragma unroll
    for (int j = 0; j < 8; ++j) acco[t][j] = 0.f;
  const v8f zero8 = {0.f, 0.f, 0.f, 0.f, 0.f, 0.f, 0.f, 0.f};

  __bf16* pw = &pbuf[wv * 32 * 16];

  for (int m0 = 0; m0 < N; m0 += 32) {
    // K fragments for two 16-key tiles (B layout: lane-half = channel range)
    v16bf kf0, kf1;
    #pragma unroll
    for (int i = 0; i < 16; ++i) {
      const int cch = ((i >> 3) << 4) | (hi << 3) | (i & 7);
      kf0[i] = (__bf16)kh[(size_t)cch * N + m0 + col];
      kf1[i] = (__bf16)kh[(size_t)cch * N + m0 + 16 + col];
    }
    v8f s0 = WMMA_BF16(qf, kf0, zero8);
    v8f s1 = WMMA_BF16(qf, kf1, zero8);

    #pragma unroll
    for (int j = 0; j < 8; ++j) {
      float a = s0[j], c2 = s1[j];
      float mx = fmaxf(a, c2);
      mx = fmaxf(mx, __shfl_xor(mx, 1, 32));
      mx = fmaxf(mx, __shfl_xor(mx, 2, 32));
      mx = fmaxf(mx, __shfl_xor(mx, 4, 32));
      mx = fmaxf(mx, __shfl_xor(mx, 8, 32));
      const float mnew = fmaxf(mrun[j], mx);
      const float corr = __expf(mrun[j] - mnew);
      const float p0 = __expf(a - mnew);
      const float p1 = __expf(c2 - mnew);
      float ps = p0 + p1;
      ps += __shfl_xor(ps, 1, 32);
      ps += __shfl_xor(ps, 2, 32);
      ps += __shfl_xor(ps, 4, 32);
      ps += __shfl_xor(ps, 8, 32);
      lrun[j] = lrun[j] * corr + ps;
      mrun[j] = mnew;
      acco[0][j] *= corr; acco[1][j] *= corr; acco[2][j] *= corr; acco[3][j] *= corr;

      // scatter P into A-fragment order (row = query, k = key-in-32)
      const int rr  = j + hi * 8;
      const int ml0 = col, ml1 = 16 + col;
      pw[((ml0 >> 4) * 16 + rr) * 16 + (ml0 & 15)] = (__bf16)p0;
      pw[((ml1 >> 4) * 16 + rr) * 16 + (ml1 & 15)] = (__bf16)p1;
    }
    __syncthreads();

    const v16bf pf = *reinterpret_cast<const v16bf*>(&pw[lane * 16]);
    #pragma unroll
    for (int t = 0; t < 4; ++t) {
      const int d = t * 16 + col;
      const float* vr = vh + (size_t)d * N + m0 + hi * 16;   // 64B aligned
      float4 f0 = reinterpret_cast<const float4*>(vr)[0];
      float4 f1 = reinterpret_cast<const float4*>(vr)[1];
      float4 f2 = reinterpret_cast<const float4*>(vr)[2];
      float4 f3 = reinterpret_cast<const float4*>(vr)[3];
      v16bf vf;
      vf[0]=(__bf16)f0.x; vf[1]=(__bf16)f0.y; vf[2]=(__bf16)f0.z; vf[3]=(__bf16)f0.w;
      vf[4]=(__bf16)f1.x; vf[5]=(__bf16)f1.y; vf[6]=(__bf16)f1.z; vf[7]=(__bf16)f1.w;
      vf[8]=(__bf16)f2.x; vf[9]=(__bf16)f2.y; vf[10]=(__bf16)f2.z; vf[11]=(__bf16)f2.w;
      vf[12]=(__bf16)f3.x; vf[13]=(__bf16)f3.y; vf[14]=(__bf16)f3.z; vf[15]=(__bf16)f3.w;
      acco[t] = WMMA_BF16(pf, vf, acco[t]);
    }
    __syncthreads();
  }

  #pragma unroll
  for (int t = 0; t < 4; ++t)
    #pragma unroll
    for (int j = 0; j < 8; ++j) {
      const int nn = n0 + j + hi * 8;
      const int d  = t * 16 + col;
      aout[((size_t)b * C + h * HD + d) * N + nn] = acco[t][j] / lrun[j];
    }
}

// ---------------------------------------------------------------------------
// pe: aout += affine(dwconv3x3(v_img)). One block per (b,c); the 32x32 v
// image is staged to LDS once via async b128 copies (1 per lane).
// ---------------------------------------------------------------------------
__global__ __launch_bounds__(256)
void pe_kernel(const float* __restrict__ qkv, const float* __restrict__ pw,
               const float* __restrict__ ps, const float* __restrict__ pb,
               float* __restrict__ aout)
{
  __shared__ __align__(16) float tile[1024];
  const int bc = blockIdx.x;
  const int c = bc & 255, b = bc >> 8;
  const int t = threadIdx.x;
  const int head = c >> 6, dd = c & 63;
  const float* vimg = qkv + ((size_t)b * HQKV + head * 128 + 2 * KD + dd) * N;

  async_load_b128(&tile[t * 4], vimg + t * 4);

  float w9[9];
  #pragma unroll
  for (int i = 0; i < 9; ++i) w9[i] = pw[c * 9 + i];
  const float sc = ps[c], bi = pb[c];
  float* arow = aout + ((size_t)b * C + c) * N;

  async_wait0();
  __syncthreads();

  #pragma unroll
  for (int r = 0; r < 4; ++r) {
    const int n = t + r * 256;
    const int hh = n >> 5, ww = n & 31;
    float acc = 0.f;
    #pragma unroll
    for (int dy = -1; dy <= 1; ++dy)
      #pragma unroll
      for (int dx = -1; dx <= 1; ++dx) {
        const int yy = hh + dy, xx = ww + dx;
        if (yy >= 0 && yy < 32 && xx >= 0 && xx < 32)
          acc += tile[yy * 32 + xx] * w9[(dy + 1) * 3 + (dx + 1)];
      }
    arow[n] += acc * sc + bi;
  }
}

// ---------------------------------------------------------------------------
// depthwise 3x3 on 1024 ch + exact-GELU gate -> 512 ch. One block per
// (b, gate-pair); both channel images async-staged to LDS.
// ---------------------------------------------------------------------------
__global__ __launch_bounds__(256)
void dwgelu_kernel(const float* __restrict__ y, const float* __restrict__ dww,
                   float* __restrict__ z)
{
  __shared__ __align__(16) float t1[1024];
  __shared__ __align__(16) float t2[1024];
  const int bc = blockIdx.x;
  const int c = bc & 511, b = bc >> 9;
  const int t = threadIdx.x;
  const float* y1 = y + ((size_t)b * 1024 + c) * N;
  const float* y2 = y + ((size_t)b * 1024 + c + 512) * N;

  async_load_b128(&t1[t * 4], y1 + t * 4);
  async_load_b128(&t2[t * 4], y2 + t * 4);

  float wa[9], wb[9];
  #pragma unroll
  for (int i = 0; i < 9; ++i) { wa[i] = dww[c * 9 + i]; wb[i] = dww[(c + 512) * 9 + i]; }
  float* zrow = z + ((size_t)b * HID + c) * N;

  async_wait0();
  __syncthreads();

  #pragma unroll
  for (int r = 0; r < 4; ++r) {
    const int n = t + r * 256;
    const int hh = n >> 5, ww = n & 31;
    float d1 = 0.f, d2 = 0.f;
    #pragma unroll
    for (int dy = -1; dy <= 1; ++dy)
      #pragma unroll
      for (int dx = -1; dx <= 1; ++dx) {
        const int yy = hh + dy, xx = ww + dx;
        if (yy >= 0 && yy < 32 && xx >= 0 && xx < 32) {
          const int o = yy * 32 + xx, k = (dy + 1) * 3 + (dx + 1);
          d1 += t1[o] * wa[k];
          d2 += t2[o] * wb[k];
        }
      }
    const float g = 0.5f * d1 * (1.f + erff(d1 * 0.70710678118654752f));
    zrow[n] = g * d2;
  }
}

// ---------------------------------------------------------------------------
// k = irfft2(filt) per channel: real filter => spatial kernel via 8-pt cosines
// ---------------------------------------------------------------------------
__global__ void kfilt_kernel(const float* __restrict__ filt, float* __restrict__ ks)
{
  const int c = blockIdx.x, t = threadIdx.x;
  const int p = t >> 3, q = t & 7;
  const float ctab[8] = {1.f, 0.70710678118f, 0.f, -0.70710678118f,
                         -1.f, -0.70710678118f, 0.f, 0.70710678118f};
  float s = 0.f;
  for (int u = 0; u < 8; ++u)
    for (int v = 0; v < 8; ++v) {
      const float F = (v <= 4) ? filt[c * 40 + u * 5 + v]
                               : filt[c * 40 + ((8 - u) & 7) * 5 + (8 - v)];
      s += F * ctab[(u * p + v * q) & 7];
    }
  ks[c * 64 + t] = s * (1.f / 64.f);
}

// ---------------------------------------------------------------------------
// out = x2 + circular-conv8x8(y3, ks)  (== irfft2(rfft2(y3)*filt) per block).
// One block per (b,c); channel image + 8x8 kernel async-staged to LDS
// (each 8x8 block would otherwise be re-read 64x from global).
// ---------------------------------------------------------------------------
__global__ __launch_bounds__(256)
void circconv_kernel(const float* __restrict__ y3, const float* __restrict__ ks,
                     const float* __restrict__ x2, float* __restrict__ out)
{
  __shared__ __align__(16) float img[1024];
  __shared__ __align__(16) float kern[64];
  const int bc = blockIdx.x;
  const int c = bc & 255, b = bc >> 8;
  const int t = threadIdx.x;
  const float* yb = y3 + ((size_t)b * C + c) * N;
  const float* kc = ks + c * 64;

  async_load_b128(&img[t * 4], yb + t * 4);
  if (t < 16) async_load_b128(&kern[t * 4], kc + t * 4);
  async_wait0();
  __syncthreads();

  const size_t rowo = ((size_t)b * C + c) * N;
  #pragma unroll
  for (int r = 0; r < 4; ++r) {
    const int n = t + r * 256;
    const int hh = n >> 5, ww = n & 31;
    const int hb = hh & ~7, wb = ww & ~7, p = hh & 7, q = ww & 7;
    float acc = 0.f;
    #pragma unroll
    for (int a = 0; a < 8; ++a)
      #pragma unroll
      for (int e = 0; e < 8; ++e)
        acc += img[(hb + a) * 32 + wb + e] * kern[((p - a) & 7) * 8 + ((q - e) & 7)];
    out[rowo + n] = x2[rowo + n] + acc;
  }
}

// ---------------------------------------------------------------------------
extern "C" void kernel_launch(void* const* d_in, const int* in_sizes, int n_in,
                              void* d_out, int out_size, void* d_ws, size_t ws_size,
                              hipStream_t stream)
{
  (void)in_sizes; (void)n_in; (void)out_size; (void)ws_size;
  const float* x      = (const float*)d_in[0];
  const float* qkv_w  = (const float*)d_in[1];
  const float* qkv_s  = (const float*)d_in[2];
  const float* qkv_b  = (const float*)d_in[3];
  const float* pe_w   = (const float*)d_in[4];
  const float* pe_s   = (const float*)d_in[5];
  const float* pe_b   = (const float*)d_in[6];
  const float* proj_w = (const float*)d_in[7];
  const float* proj_s = (const float*)d_in[8];
  const float* proj_b = (const float*)d_in[9];
  const float* pin_w  = (const float*)d_in[10];
  const float* dw_w   = (const float*)d_in[11];
  const float* fftf   = (const float*)d_in[12];
  const float* pout_w = (const float*)d_in[13];
  float* out = (float*)d_out;

  // workspace layout (floats); total ~46.2M floats (~185 MB)
  float* ws   = (float*)d_ws;
  float* qkv  = ws;                    // 16*512*1024
  float* aout = ws + 8388608;          // 16*256*1024  (becomes out+pe in place)
  float* x2   = ws + 12582912;         // 16*256*1024
  float* y    = ws + 16777216;         // 16*1024*1024
  float* z    = ws + 33554432;         // 16*512*1024
  float* y3   = ws + 41943040;         // 16*256*1024
  float* ks   = ws + 46137344;         // 256*64

  const dim3 blk(256);

  // 1) qkv = affine(x @ qkv_w^T)
  gemm_bf16_kernel<true, false><<<dim3(8, HQKV / 64, Bb), blk, 0, stream>>>(
      qkv_w, x, qkv, qkv_s, qkv_b, nullptr, HQKV, C);

  // 2) flash attention -> aout [b, C, n]
  attn_kernel<<<dim3(8, NH, Bb), blk, 0, stream>>>(qkv, aout);

  // 3) aout += affine(dwconv3x3(v))
  pe_kernel<<<dim3(Bb * C), blk, 0, stream>>>(qkv, pe_w, pe_s, pe_b, aout);

  // 4) x2 = x + affine(aout @ proj_w^T)
  gemm_bf16_kernel<true, true><<<dim3(8, C / 64, Bb), blk, 0, stream>>>(
      proj_w, aout, x2, proj_s, proj_b, x, C, C);

  // 5) y = x2 @ pin_w^T  (1024 ch)
  gemm_bf16_kernel<false, false><<<dim3(8, (2 * HID) / 64, Bb), blk, 0, stream>>>(
      pin_w, x2, y, nullptr, nullptr, nullptr, 2 * HID, C);

  // 6) z = gelu(dw(y)[:512]) * dw(y)[512:]
  dwgelu_kernel<<<dim3(Bb * HID), blk, 0, stream>>>(y, dw_w, z);

  // 7) y3 = z @ pout_w^T
  gemm_bf16_kernel<false, false><<<dim3(8, C / 64, Bb), blk, 0, stream>>>(
      pout_w, z, y3, nullptr, nullptr, nullptr, C, HID);

  // 8) spatial kernel from FFT filter, then out = x2 + circconv(y3)
  kfilt_kernel<<<dim3(C), dim3(64), 0, stream>>>(fftf, ks);
  circconv_kernel<<<dim3(Bb * C), blk, 0, stream>>>(y3, ks, x2, out);
}
